// MSAEncoder_71794673320039
// MI455X (gfx1250) — matrix operand order repaired and checked by reading, not compiled, not run on hardware
//
#include <hip/hip_runtime.h>
#include <hip/hip_bf16.h>
#include <math.h>

// MSA covariance encoder for MI455X (gfx1250, wave32, WMMA).
//
// Design: per-edge 21x21 weighted joint histogram expressed as a
// (21x200)x(200x21) GEMM on v_wmma_f32_16x16x32_f16: A holds w[k]-weighted
// one-hots of x[i,:], B holds 0/1 one-hots of x[j,:]. 2x2 tiles of 16x16,
// K padded 200->224 (7 chunks of 32). One wave per edge, EXEC all-ones.
// Workload is write-BW bound (~35 MB out => ~1.5us at 23.3 TB/s).

typedef __attribute__((ext_vector_type(16))) _Float16 v16h;
typedef __attribute__((ext_vector_type(8)))  float    v8f;

#define NSPEC   200
#define KPAD    224     // 7 * 32
#define AA      21
#define WPB     8       // waves per block (256 threads)

// ---------------------------------------------------------------- softmax(W)
__global__ void k_softmax(const float* __restrict__ W,
                          float* __restrict__ w_f32,
                          _Float16* __restrict__ w_f16) {
    __shared__ float red[256];
    int t = threadIdx.x;
    float v = (t < NSPEC) ? W[t] : -3.4e38f;
    red[t] = v; __syncthreads();
    for (int s = 128; s > 0; s >>= 1) {
        if (t < s) red[t] = fmaxf(red[t], red[t + s]);
        __syncthreads();
    }
    float mx = red[0]; __syncthreads();
    float e = (t < NSPEC) ? __expf(v - mx) : 0.f;
    red[t] = e; __syncthreads();
    for (int s = 128; s > 0; s >>= 1) {
        if (t < s) red[t] += red[t + s];
        __syncthreads();
    }
    float inv = 1.0f / red[0];
    if (t < KPAD) {
        float wv = (t < NSPEC) ? e * inv : 0.f;
        w_f32[t] = wv;
        w_f16[t] = (_Float16)wv;
    }
}

// ------------------------------------------------- x1[l,a] = sum_n w[n]*1[x==a]
__global__ void k_x1(const int* __restrict__ x,
                     const float* __restrict__ w,
                     float* __restrict__ x1, int L) {
    __shared__ float acc[256 * AA];           // 21.5 KB of the 320 KB WGP LDS
    int t = threadIdx.x;
    int l = blockIdx.x * 256 + t;
    float* a = &acc[t * AA];
#pragma unroll
    for (int q = 0; q < AA; ++q) a[q] = 0.f;
    if (l < L) {
        const int* row = x + l * NSPEC;
        for (int n = 0; n < NSPEC; ++n) a[row[n]] += w[n];   // ds_add, per-thread bins
        float* o = x1 + l * AA;
#pragma unroll
        for (int q = 0; q < AA; ++q) o[q] = a[q];
    }
}

// -------------------------------------------------------- per-edge WMMA kernel
__global__ void __launch_bounds__(256) k_edges(const int* __restrict__ x,
                                               const int* __restrict__ ei,
                                               const _Float16* __restrict__ wf16,
                                               const float* __restrict__ x1,
                                               float* __restrict__ out2, int E) {
    __shared__ int      s_xi[WPB][KPAD];
    __shared__ int      s_xj[WPB][KPAD];
    __shared__ float    s_x1i[WPB][24];
    __shared__ float    s_x1j[WPB][24];
    __shared__ _Float16 s_w[KPAD];

    const int t    = threadIdx.x;
    const int lane = t & 31;
    const int wv   = t >> 5;
    int e = blockIdx.x * WPB + wv;
    if (e >= E) e = E - 1;            // clamp (never drop a wave: WMMA needs EXEC all-1s)

    const int i = ei[e];
    const int j = ei[E + e];

    // block-wide: stage f16 softmax weights
    for (int q = t; q < KPAD; q += 256) s_w[q] = wf16[q];
    // per-wave: stage residue ids (pad with -1 so pads match nothing)
    {
        const int* xi = x + i * NSPEC;
        const int* xj = x + j * NSPEC;
        for (int q = lane; q < KPAD; q += 32) {
            s_xi[wv][q] = (q < NSPEC) ? xi[q] : -1;
            s_xj[wv][q] = (q < NSPEC) ? xj[q] : -1;
        }
        if (lane < AA) {
            s_x1i[wv][lane] = x1[i * AA + lane];
            s_x1j[wv][lane] = x1[j * AA + lane];
        }
    }
    __syncthreads();

    const int lmod = lane & 15;       // A: row within tile; B/C: column within tile
    const int lhi  = lane >> 4;       // lane group
    const int kloA = lhi * 8;         // A-layout K offset for lane group
    const short cz = 0;

    v8f c00 = {}, c01 = {}, c10 = {}, c11 = {};
    const _Float16 h1 = (_Float16)1.0f;
    const _Float16 h0 = (_Float16)0.0f;

    for (int kc = 0; kc < 7; ++kc) {
        const int kb = kc * 32;
        v16h a0, a1, b0, b1;
#pragma unroll
        for (int el = 0; el < 16; ++el) {
            // A 16x32 f16 layout: elem e<8 -> K = klo+e ; e>=8 -> K = 16+klo+(e-8)
            const int kkA = kb + ((el & 8) << 1) + kloA + (el & 7);
            const int xv  = s_xi[wv][kkA];
            const _Float16 wval = s_w[kkA];
            a0[el] = (xv == lmod)      ? wval : h0;   // rows  0..15
            a1[el] = (xv == lmod + 16) ? wval : h0;   // rows 16..31 (only 16..20 live)
            // B 32x16 f16 layout: lane = column n, elem e<8 -> K = klo+e ; e>=8 -> K = 16+klo+(e-8)
            const int kkB = kb + ((el & 8) << 1) + kloA + (el & 7);
            const int yv  = s_xj[wv][kkB];
            b0[el] = (yv == lmod)      ? h1 : h0;     // cols  0..15
            b1[el] = (yv == lmod + 16) ? h1 : h0;     // cols 16..31 (only 16..20 live)
        }
        c00 = __builtin_amdgcn_wmma_f32_16x16x32_f16(false, a0, false, b0, cz, c00, false, false);
        c01 = __builtin_amdgcn_wmma_f32_16x16x32_f16(false, a0, false, b1, cz, c01, false, false);
        c10 = __builtin_amdgcn_wmma_f32_16x16x32_f16(false, a1, false, b0, cz, c10, false, false);
        c11 = __builtin_amdgcn_wmma_f32_16x16x32_f16(false, a1, false, b1, cz, c11, false, false);
    }

    // Epilogue: covariance subtraction, squared-norm accumulation, scatter-out.
    // C layout: VGPR r -> m = r (lanes 0-15) or r+8 (lanes 16-31); lane -> n.
    float sumsq = 0.f;
    float* oe = out2 + (size_t)e * (AA * AA + 1);
    const float* x1iL = s_x1i[wv];
    const float* x1jL = s_x1j[wv];

    auto emit = [&](const v8f& c, int tr, int tc) {
        const int n = tc * 16 + lmod;
#pragma unroll
        for (int r = 0; r < 8; ++r) {
            const int m = tr * 16 + lhi * 8 + r;
            if (m < AA && n < AA) {
                float v = c[r] - x1iL[m] * x1jL[n];
                sumsq += v * v;
                oe[m * AA + n] = v;
            }
        }
    };
    emit(c00, 0, 0); emit(c01, 0, 1); emit(c10, 1, 0); emit(c11, 1, 1);

    // wave32 reduction of the squared norm
    for (int off = 16; off > 0; off >>= 1)
        sumsq += __shfl_xor(sumsq, off, 32);
    if (lane == 0)
        oe[AA * AA] = sqrtf(sumsq + 1e-12f);
}

// ------------------------------------------------------------------- launcher
extern "C" void kernel_launch(void* const* d_in, const int* in_sizes, int n_in,
                              void* d_out, int out_size, void* d_ws, size_t ws_size,
                              hipStream_t stream) {
    const int*   x  = (const int*)d_in[0];    // (L, 200) residue ids
    const int*   ei = (const int*)d_in[1];    // (2, E) edge indices
    const float* W  = (const float*)d_in[2];  // (1, 200) species logits

    const int L = in_sizes[0] / NSPEC;
    const int E = in_sizes[1] / 2;

    float* out  = (float*)d_out;
    float* x1   = out;                        // L*21
    float* out2 = out + (size_t)L * AA;       // E*442

    float*    w_f32 = (float*)d_ws;
    _Float16* w_f16 = (_Float16*)((char*)d_ws + KPAD * sizeof(float));

    k_softmax<<<1, 256, 0, stream>>>(W, w_f32, w_f16);
    k_x1<<<(L + 255) / 256, 256, 0, stream>>>(x, w_f32, x1, L);
    k_edges<<<(E + WPB - 1) / WPB, 256, 0, stream>>>(x, ei, w_f16, x1, out2, E);
}
